// Ligand_SE3_18580028522894
// MI455X (gfx1250) — matrix operand order repaired and checked by reading, not compiled, not run on hardware
//
// SE(3)-Transformer forward for MI455X (gfx1250). Fused radial-GEMM + TFN conv
// via v_wmma_f32_16x16x32_bf16 with pre-packed bf16 B-operands + padded bias;
// wave-local s_wait_dscnt sync; global_prefetch for next B block; attention on
// VALU with atomics.
#include <hip/hip_runtime.h>
#include <cmath>
#include <complex>
#include <vector>
#include <cstdint>

#define N_NODES 4096
#define N_EDGES 16384
#define L0_IN   15
#define KV_C    8
#define L0_OUT  32
#define PI_F    3.14159265358979323846f

typedef __bf16 bf16_t;
typedef __attribute__((ext_vector_type(16))) __bf16 v16bf;
typedef __attribute__((ext_vector_type(8)))  float  v8f;

// ---------------- constant index tables (CG & basis layouts) ----------------
__constant__ int c_cgOff[3][3][5] = {
  { {0,0,0,0,0},      {1,0,0,0,0},       {10,0,0,0,0} },
  { {35,0,0,0,0},     {44,53,80,0,0},    {125,170,245,0,0} },
  { {350,0,0,0,0},    {375,420,495,0,0}, {600,625,700,825,1000} } };
__constant__ int c_basisPairOff[3][3] = { {0,1,4}, {9,12,39}, {84,89,134} };

// ---------------- workspace layout (float elements) ----------------
constexpr size_t WS_CG    = 0;                                   // 1225 (pad 1280)
constexpr size_t WS_Y     = 1280;                                // E*25
constexpr size_t WS_ESC   = WS_Y    + (size_t)N_EDGES*25;
constexpr size_t WS_BASIS = WS_ESC  + (size_t)N_EDGES*6;         // E*259
constexpr size_t WS_H2A   = WS_BASIS+ (size_t)N_EDGES*259;
constexpr size_t WS_H2B   = WS_H2A  + (size_t)N_EDGES*32;
constexpr size_t WS_H2C   = WS_H2B  + (size_t)N_EDGES*32;
constexpr size_t WS_KV1   = WS_H2C  + (size_t)N_EDGES*32;        // E*16*9
constexpr size_t WS_KV2   = WS_KV1  + (size_t)N_EDGES*16*9;
constexpr size_t WS_Q1    = WS_KV2  + (size_t)N_EDGES*16*9;      // N*8
constexpr size_t WS_Q2    = WS_Q1   + (size_t)N_NODES*8;         // N*8*9
constexpr size_t WS_LOGIT = WS_Q2   + (size_t)N_NODES*8*9;       // E*4
constexpr size_t WS_ZMAX  = WS_LOGIT+ (size_t)N_EDGES*4;         // N*4 (uint)
constexpr size_t WS_AEXP  = WS_ZMAX + (size_t)N_NODES*4;         // E*4
constexpr size_t WS_DENOM = WS_AEXP + (size_t)N_EDGES*4;         // N*4
constexpr size_t WS_AGG1  = WS_DENOM+ (size_t)N_NODES*4;         // N*8*9
constexpr size_t WS_AGG2  = WS_AGG1 + (size_t)N_NODES*8*9;
constexpr size_t WS_F1    = WS_AGG2 + (size_t)N_NODES*8*9;       // N*32*9
constexpr size_t WS_F2    = WS_F1   + (size_t)N_NODES*32*9;
// packed bf16 W3 copies (column-major per padded column, 32 K values each)
constexpr size_t WS_W3P1  = WS_F2   + (size_t)N_NODES*32*9;      // 768*32 bf16  = 12288 f
constexpr size_t WS_W3P2  = WS_W3P1 + 12288;                     // 9728*32 bf16 = 155648 f
constexpr size_t WS_W3PF  = WS_W3P2 + 155648;                    // 3072*32 bf16 = 49152 f
// zero-padded bias copies (padded column space)
constexpr size_t WS_B3P1  = WS_W3PF + 49152;                     // 768 f
constexpr size_t WS_B3P2  = WS_B3P1 + 768;                       // 9728 f
constexpr size_t WS_B3PF  = WS_B3P2 + 9728;                      // 3072 f
constexpr size_t WS_END   = WS_B3PF + 3072;                      // ~59 MB total

// ---------------- host: build real Clebsch-Gordan table (load-time const) ---
static double dfact(int n){ double r=1; for(int i=2;i<=n;++i) r*=(double)i; return r; }
static double cg_coef(int j1,int m1,int j2,int m2,int j3,int m3){
  if (m1+m2!=m3) return 0.0;
  double pref = std::sqrt((2*j3+1)*dfact(j3+j1-j2)*dfact(j3-j1+j2)*dfact(j1+j2-j3)/dfact(j1+j2+j3+1));
  pref *= std::sqrt(dfact(j3+m3)*dfact(j3-m3)*dfact(j1-m1)*dfact(j1+m1)*dfact(j2-m2)*dfact(j2+m2));
  double s=0.0;
  for(int k=0;k<=j1+j2-j3;++k){
    int d0=k,d1=j1+j2-j3-k,d2=j1-m1-k,d3=j2+m2-k,d4=j3-j2+m1+k,d5=j3-j1-m2+k;
    if(d0<0||d1<0||d2<0||d3<0||d4<0||d5<0) continue;
    double den=dfact(d0)*dfact(d1)*dfact(d2)*dfact(d3)*dfact(d4)*dfact(d5);
    s += ((k&1)?-1.0:1.0)/den;
  }
  return pref*s;
}
static void u_real_mat(int l, std::complex<double>* U){
  int n=2*l+1;
  for(int i=0;i<n*n;++i) U[i]=0.0;
  U[l*n+l]=1.0;
  const double is2 = 1.0/std::sqrt(2.0);
  for(int m=1;m<=l;++m){
    double sgn = (m&1)?-1.0:1.0;
    U[(l+m)*n+(l+m)] = sgn*is2;
    U[(l+m)*n+(l-m)] = is2;
    U[(l-m)*n+(l-m)] = std::complex<double>(0.0,is2);
    U[(l-m)*n+(l+m)] = std::complex<double>(0.0,-sgn*is2);
  }
}
static void real_cg(int j1,int j2,int j3, float* out){
  int n1=2*j1+1,n2=2*j2+1,n3=2*j3+1;
  std::vector<std::complex<double>> U1(n1*n1),U2(n2*n2),U3(n3*n3),T((size_t)n1*n2*n3,0.0);
  u_real_mat(j1,U1.data()); u_real_mat(j2,U2.data()); u_real_mat(j3,U3.data());
  for(int a=0;a<n1;++a)for(int b=0;b<n2;++b)for(int c=0;c<n3;++c){
    std::complex<double> acc=0.0;
    for(int u=0;u<n1;++u){
      std::complex<double> Au=U1[a*n1+u]; if(Au==std::complex<double>(0.0)) continue;
      for(int v=0;v<n2;++v){
        std::complex<double> Bv=U2[b*n2+v]; if(Bv==std::complex<double>(0.0)) continue;
        int m1=u-j1,m2=v-j2,m3=m1+m2;
        if(m3<-j3||m3>j3) continue;
        std::complex<double> Cw=std::conj(U3[c*n3+(m3+j3)]);
        if(Cw==std::complex<double>(0.0)) continue;
        acc += Au*Bv*Cw*cg_coef(j1,m1,j2,m2,j3,m3);
      }
    }
    T[((size_t)a*n2+b)*n3+c]=acc;
  }
  double nr=0,ni=0;
  for(auto&z:T){ nr+=z.real()*z.real(); ni+=z.imag()*z.imag(); }
  bool useR = nr>=ni;
  for(size_t i=0;i<T.size();++i) out[i] = (float)(useR? T[i].real() : T[i].imag());
}
struct CGTable {
  float v[1280];
  CGTable(){
    for(int i=0;i<1280;++i) v[i]=0.f;
    int off=0;
    for(int di=0;di<3;++di) for(int dd=0;dd<3;++dd){
      int Jmin = di>dd? di-dd : dd-di;
      for(int J=Jmin;J<=di+dd;++J){
        real_cg(di,J,dd, v+off);
        off += (2*di+1)*(2*J+1)*(2*dd+1);
      }
    }
  }
};
static const CGTable g_cg;   // pure constant; built once at library load

// ---------------- device helpers ----------------
__device__ __forceinline__ unsigned omap(float f){
  unsigned u=__float_as_uint(f); return (u&0x80000000u)? ~u : (u|0x80000000u);
}
__device__ __forceinline__ float oinv(unsigned u){
  return (u&0x80000000u)? __uint_as_float(u^0x80000000u) : __uint_as_float(~u);
}

// ---------------- kernels ----------------
// 1) per-edge geometry: spherical harmonics Y_0..Y_4 (25) and edge scalars (6)
__global__ void geom_kernel(const float* __restrict__ pos, const float* __restrict__ eattr,
                            const int* __restrict__ src, const int* __restrict__ dst,
                            float* __restrict__ Y, float* __restrict__ esc){
  int e = blockIdx.x*blockDim.x + threadIdx.x;
  if(e>=N_EDGES) return;
  int s=src[e], t=dst[e];
  float x=pos[t*3+0]-pos[s*3+0];
  float y=pos[t*3+1]-pos[s*3+1];
  float z=pos[t*3+2]-pos[s*3+2];
  float r = sqrtf(x*x+y*y+z*z+1e-12f);
  float ct = z/r;
  float st = sqrtf(x*x+y*y)/r;
  float phi = atan2f(y,x);
  float P[5][5];
  P[0][0]=1.f;
  for(int m=1;m<=4;++m) P[m][m]=P[m-1][m-1]*st*(float)(2*m-1);
  for(int m=0;m<4;++m)  P[m+1][m]=ct*(float)(2*m+1)*P[m][m];
  for(int m=0;m<=4;++m)
    for(int l=m+2;l<=4;++l)
      P[l][m]=((float)(2*l-1)*ct*P[l-1][m]-(float)(l+m-1)*P[l-2][m])/(float)(l-m);
  const float fct[9]={1.f,1.f,2.f,6.f,24.f,120.f,720.f,5040.f,40320.f};
  float* ye = Y + (size_t)e*25;
  for(int l=0;l<=4;++l){
    for(int m=-l;m<=l;++m){
      int am = m<0?-m:m;
      float Nlm = sqrtf((float)(2*l+1)/(4.f*PI_F)*fct[l-am]/fct[l+am]);
      float vv;
      if(m==0)      vv = Nlm*P[l][0];
      else if(m>0)  vv = 1.41421356237f*Nlm*P[l][am]*cosf((float)am*phi);
      else          vv = 1.41421356237f*Nlm*P[l][am]*sinf((float)am*phi);
      ye[l*l + (m+l)] = vv;
    }
  }
  for(int i=0;i<5;++i) esc[e*6+i]=eattr[e*5+i];
  esc[e*6+5]=r;
}

// 2) per-edge CG basis: basis[e,(din,dout),J,o,i] = sum_m CG[i,m,o]*Y_J[m]
__global__ __launch_bounds__(128)
void basis_kernel(const float* __restrict__ Y, const float* __restrict__ cg,
                  float* __restrict__ basis){
  __shared__ float scg[1232];
  for(int i=threadIdx.x;i<1225;i+=128) scg[i]=cg[i];
  __syncthreads();
  int e = blockIdx.x*128 + threadIdx.x;
  if(e>=N_EDGES) return;
  const float* ye = Y + (size_t)e*25;
  float* be = basis + (size_t)e*259;
  for(int di=0;di<3;++di){
    int ndi=2*di+1;
    for(int dd=0;dd<3;++dd){
      int ndo=2*dd+1;
      int mn = di<dd?di:dd;
      int Jmin = di>dd? di-dd : dd-di;
      int nJ = 2*mn+1;
      int poff = c_basisPairOff[di][dd];
      for(int jj=0;jj<nJ;++jj){
        int J=Jmin+jj, nm=2*J+1;
        const float* C = scg + c_cgOff[di][dd][jj];
        for(int o=0;o<ndo;++o)
          for(int i=0;i<ndi;++i){
            float acc=0.f;
            for(int m=0;m<nm;++m) acc += C[(i*nm+m)*ndo+o]*ye[J*J+m];
            be[poff + (jj*ndo+o)*ndi + i] = acc;
          }
      }
    }
  }
}

// 3) radial MLP hidden: h2 = relu(relu(esc@w1+b1)@w2+b2)   (E x 32)
__global__ __launch_bounds__(256)
void radial_hidden_kernel(const float* __restrict__ esc,
                          const float* __restrict__ w1, const float* __restrict__ b1,
                          const float* __restrict__ w2, const float* __restrict__ b2,
                          float* __restrict__ h2out){
  __shared__ float sw1[6*32], sw2[32*32], sb1[32], sb2[32];
  int tid=threadIdx.x;
  for(int i=tid;i<192;i+=256)  sw1[i]=w1[i];
  for(int i=tid;i<1024;i+=256) sw2[i]=w2[i];
  if(tid<32){ sb1[tid]=b1[tid]; sb2[tid]=b2[tid]; }
  __syncthreads();
  int e = blockIdx.x*256 + tid;
  if(e>=N_EDGES) return;
  float x[6];
  for(int i=0;i<6;++i) x[i]=esc[e*6+i];
  float h1[32];
  for(int o=0;o<32;++o){
    float a=sb1[o];
    for(int i=0;i<6;++i) a += x[i]*sw1[i*32+o];
    h1[o]=fmaxf(a,0.f);
  }
  for(int o=0;o<32;++o){
    float a=sb2[o];
    for(int k=0;k<32;++k) a += h1[k]*sw2[k*32+o];
    h2out[(size_t)e*32+o]=fmaxf(a,0.f);
  }
}

// 3b) pack W3 (32 x outdim, f32 row-major) into bf16, column-major per padded
//     column: w3p[colP*32 + k]; also zero-padded bias b3p[colP].
//     Fragment-ready: a lane's whole B-fragment = 16 consecutive bf16.
__global__ void pack_w3_kernel(const float* __restrict__ w3, const float* __restrict__ b3,
                               bf16_t* __restrict__ w3p, float* __restrict__ b3p,
                               int outdim, int cin, int cinP, int ncolP){
  int colP = blockIdx.x*blockDim.x + threadIdx.x;
  if(colP>=ncolP) return;
  int blk = colP/cinP, cc = colP - blk*cinP;
  bf16_t* dp = w3p + (size_t)colP*32;
  if(cc < cin){
    int col = blk*cin + cc;
    for(int k=0;k<32;++k) dp[k] = (bf16_t)w3[(size_t)k*outdim + col];
    b3p[colP] = b3[col];
  } else {
    for(int k=0;k<32;++k) dp[k] = (bf16_t)0.f;
    b3p[colP] = 0.f;
  }
}

// 4) fused: R-block GEMM (WMMA bf16->f32) + TFN conv contraction.
//    One block = 16 edges, 4 waves; wave w owns output channels d = w, w+4, ...
//    s_R is wave-private: WMMA D -> LDS -> consume synced with s_wait_dscnt.
__global__ __launch_bounds__(128)
void fused_conv_kernel(const float* __restrict__ h2, const bf16_t* __restrict__ w3p,
                       const float* __restrict__ b3p, const float* __restrict__ basis,
                       const float* __restrict__ fp0, const float* __restrict__ fp1,
                       const float* __restrict__ fp2,
                       const int* __restrict__ src, const int* __restrict__ dst,
                       float* __restrict__ o0, float* __restrict__ o1, float* __restrict__ o2,
                       int n_din, int cin, int cinP, int n_dout, int cout,
                       int ncolP, int atomic_mode){
  __shared__ float s_h2[16*32];
  __shared__ float s_fe[16*32*5];
  __shared__ float s_t [16*32*5];          // layout [o][e][cP]
  __shared__ float s_R [4][16*16];         // per-wave R staging (+bias)
  __shared__ float s_out[16*16*9];

  const int tid=threadIdx.x, wave=tid>>5, lane=tid&31;
  const int e0 = blockIdx.x*16;
  const int osum = n_dout*n_dout;          // sum of (2d+1), d<n_dout
  const int outTile = 16*cout*osum;

  for(int i=tid;i<16*32;i+=128) s_h2[i]=h2[(size_t)(e0+(i>>5))*32 + (i&31)];
  for(int i=tid;i<outTile;i+=128) s_out[i]=0.f;
  __syncthreads();

  // A fragment (16 edges x K=32, bf16).  contiguous k order: k = 16*(lane/16) + s.
  v16bf afrag;
  {
    const int m = lane & 15;
    const int kadd = (lane>=16)?16:0;
    #pragma unroll
    for(int v=0;v<8;++v){
      int kb = 2*v + kadd;
      afrag[2*v]   = (bf16_t)s_h2[m*32 + kb];
      afrag[2*v+1] = (bf16_t)s_h2[m*32 + kb + 1];
    }
  }
  const int n    = lane & 15;
  const int kbase= (lane>=16)?16:0;
  const int madd = (lane>=16)?8:0;
  const int eL   = lane & 15;
  const int oh   = lane >> 4;

  int rblk=0;   // R block counter; packed colbase = blk*cinP
  for(int din=0; din<n_din; ++din){
    const int ndi=2*din+1;
    const float* fp = (din==0)?fp0:((din==1)?fp1:fp2);
    __syncthreads();
    for(int i=tid;i<16*cin*ndi;i+=128){
      int e=i/(cin*ndi); int r=i-e*(cin*ndi); int c=r/ndi; int ii=r-c*ndi;
      s_fe[i]=fp[((size_t)src[e0+e]*cin + c)*ndi + ii];
    }
    __syncthreads();
    int doutoff=0;
    for(int dout=0; dout<n_dout; ++dout){
      const int ndo=2*dout+1;
      const int nJ = 2*((din<dout)?din:dout)+1;
      const int poff = c_basisPairOff[din][dout];
      for(int jj=0;jj<nJ;++jj){
        __syncthreads();
        // t[o][e][c] = sum_i fe[e,c,i] * B[e,jj,o,i]; zero-padded c>=cin
        for(int i=tid;i<16*cinP*ndo;i+=128){
          int o=i/(16*cinP); int rem=i-o*16*cinP; int e=rem/cinP; int c=rem-e*cinP;
          float acc=0.f;
          if(c<cin){
            const float* bp = basis + (size_t)(e0+e)*259 + poff + (jj*ndo+o)*ndi;
            const float* fe = s_fe + (e*cin + c)*ndi;
            for(int ii=0;ii<ndi;++ii) acc += fe[ii]*bp[ii];
          }
          s_t[i]=acc;
        }
        __syncthreads();
        for(int d=wave; d<cout; d+=4){
          const int blk = rblk + jj*cout + d;
          const int colbaseP = blk*cinP;
          // prefetch next d-block's packed fragments (uniform scalar guard)
          {
            const int pfCol = (blk+4)*cinP;
            if(pfCol < ncolP)
              __builtin_prefetch(w3p + (size_t)pfCol*32, 0, 1);
          }
          for(int c0=0;c0<cinP;c0+=16){
            const int cc = c0 + n;
            // B fragment: one contiguous 32-byte bf16 load per lane
            v16bf bfrag = *reinterpret_cast<const v16bf*>(
                              w3p + ((size_t)(colbaseP + cc))*32 + kbase);
            v8f dfrag = {0.f,0.f,0.f,0.f,0.f,0.f,0.f,0.f};
            dfrag = __builtin_amdgcn_wmma_f32_16x16x32_bf16(
                        false, afrag, false, bfrag, (short)0, dfrag, false, false);
            // D (+padded bias, unconditional load) -> wave-private LDS tile
            const float bias = b3p[colbaseP + cc];
            #pragma unroll
            for(int p=0;p<8;++p) s_R[wave][(p+madd)*16 + n] = dfrag[p] + bias;
            // same-wave LDS RAW: split DS counter wait, no workgroup barrier
            asm volatile("s_wait_dscnt 0x0" ::: "memory");
            // out[e,d,o] += sum_c R[e,c]*t[o,e,c]   (branchless, contiguous)
            for(int o=oh;o<ndo;o+=2){
              const float* tp = s_t + ((size_t)o*16 + eL)*cinP + c0;
              const float* rp = s_R[wave] + eL*16;
              float acc=0.f;
              #pragma unroll
              for(int cp=0;cp<16;++cp) acc += rp[cp]*tp[cp];
              s_out[(eL*cout + d)*osum + doutoff + o] += acc;
            }
          }
        }
      }
      rblk   += nJ*cout;
      doutoff += ndo;
    }
  }
  __syncthreads();
  if(!atomic_mode){
    for(int i=tid;i<outTile;i+=128){
      int e=i/(cout*osum); int r=i-e*(cout*osum); int d=r/osum; int q=r-d*osum;
      int dout,o;
      if(q<1){dout=0;o=q;} else if(q<4){dout=1;o=q-1;} else {dout=2;o=q-4;}
      float* ob=(dout==0)?o0:((dout==1)?o1:o2);
      int ndo=2*dout+1;
      ob[((size_t)(e0+e)*cout + d)*ndo + o] = s_out[i];
    }
  } else {
    for(int i=tid;i<outTile;i+=128){
      int e=i/cout; int d=i-e*cout;
      atomicAdd(&o0[(size_t)dst[e0+e]*cout + d], s_out[i]);
    }
  }
}

// 5) query projection: q[n,o,m] = sum_c f[n,c,m] * wq[o,c]
__global__ void q_kernel(const float* __restrict__ f, const float* __restrict__ wq,
                         float* __restrict__ q, int cin, int nm){
  int id=blockIdx.x*blockDim.x+threadIdx.x;
  int total=N_NODES*8*nm;
  if(id>=total) return;
  int m=id%nm; int t=id/nm; int o=t%8; int n=t/8;
  float acc=0.f;
  for(int c=0;c<cin;++c) acc += f[((size_t)n*cin+c)*nm+m]*wq[o*cin+c];
  q[((size_t)n*8+o)*nm+m]=acc;
}

// 6) logits + segment-max (ordered-uint atomicMax)
__global__ void logits_kernel(const float* __restrict__ qws, const float* __restrict__ kvws,
                              const int* __restrict__ dst, float* __restrict__ logits,
                              unsigned* __restrict__ zmax, int ndeg, float inv_sqrt_dk){
  int id=blockIdx.x*blockDim.x+threadIdx.x;
  if(id>=N_EDGES*4) return;
  int h=id&3; int e=id>>2; int n=dst[e];
  float acc=0.f;
  for(int d=0;d<ndeg;++d){
    int nm=2*d+1;
    const float* q  = qws  + (size_t)N_NODES*8 *d*d;
    const float* kv = kvws + (size_t)N_EDGES*16*d*d;
    for(int cp=0;cp<2;++cp){
      int c=h*2+cp;
      for(int m=0;m<nm;++m)
        acc += q[((size_t)n*8+c)*nm+m]*kv[((size_t)e*16+c)*nm+m];
    }
  }
  acc *= inv_sqrt_dk;
  logits[e*4+h]=acc;
  atomicMax(&zmax[n*4+h], omap(acc));
}

// 7) exp and segment-sum
__global__ void expsum_kernel(const float* __restrict__ logits, const unsigned* __restrict__ zmax,
                              const int* __restrict__ dst, float* __restrict__ aexp,
                              float* __restrict__ denom){
  int id=blockIdx.x*blockDim.x+threadIdx.x;
  if(id>=N_EDGES*4) return;
  int h=id&3; int e=id>>2; int n=dst[e];
  float a=__expf(logits[id]-oinv(zmax[n*4+h]));
  aexp[id]=a;
  atomicAdd(&denom[n*4+h], a);
}

// 8) attention-weighted value aggregation (all degrees)
__global__ void agg_kernel(const float* __restrict__ kvws, const float* __restrict__ aexp,
                           const float* __restrict__ denom, const int* __restrict__ dst,
                           float* __restrict__ aggws){
  int id=blockIdx.x*blockDim.x+threadIdx.x;
  if(id>=N_EDGES*8) return;
  int c8=id&7; int e=id>>3; int h=c8>>1; int n=dst[e];
  float alpha=aexp[e*4+h]/(denom[n*4+h]+1e-9f);
  for(int d=0;d<3;++d){
    int nm=2*d+1;
    const float* kv=kvws+(size_t)N_EDGES*16*d*d;
    float* ag=aggws+(size_t)N_NODES*8*d*d;
    for(int m=0;m<nm;++m)
      atomicAdd(&ag[((size_t)n*8+c8)*nm+m], kv[((size_t)e*16+8+c8)*nm+m]*alpha);
  }
}

// 9) per-node output proj + self interaction + equivariant norm (one wave / node)
__global__ __launch_bounds__(32)
void node_update_kernel(const float* __restrict__ agg, const float* __restrict__ fprev,
                        const float* __restrict__ wo, const float* __restrict__ wsw,
                        const float* __restrict__ g, const float* __restrict__ bb,
                        float* __restrict__ fout, int d, int self_cin){
  int n=blockIdx.x; int c=threadIdx.x; int nm=2*d+1;
  float o[5];
  for(int m=0;m<nm;++m){
    float acc=0.f;
    for(int c8=0;c8<8;++c8) acc += agg[((size_t)n*8+c8)*nm+m]*wo[c*8+c8];
    if(wsw)
      for(int ci=0;ci<self_cin;++ci) acc += fprev[((size_t)n*self_cin+ci)*nm+m]*wsw[c*self_cin+ci];
    o[m]=acc;
  }
  float s2=0.f;
  for(int m=0;m<nm;++m) s2+=o[m]*o[m];
  float nv=sqrtf(s2+1e-5f);
  float s=nv;
  for(int off=16;off>=1;off>>=1) s+=__shfl_xor(s,off);
  float mu=s*(1.f/32.f);
  float dv=nv-mu;
  float vv=dv*dv;
  for(int off=16;off>=1;off>>=1) vv+=__shfl_xor(vv,off);
  float sd=sqrtf(vv*(1.f/32.f)+1e-5f);
  float nn=fmaxf(dv/sd*g[c]+bb[c],0.f);
  float sc=nn/nv;
  for(int m=0;m<nm;++m) fout[((size_t)n*32+c)*nm+m]=o[m]*sc;
}

// 10) final self term: d_out[n,o] = sum_c F2_0[n,c] * ws0[o,c]
__global__ void out_self_kernel(const float* __restrict__ f0, const float* __restrict__ ws0,
                                float* __restrict__ out){
  int id=blockIdx.x*blockDim.x+threadIdx.x;
  if(id>=N_NODES*32) return;
  int o=id&31; int n=id>>5;
  float acc=0.f;
  for(int c=0;c<32;++c) acc += f0[(size_t)n*32+c]*ws0[o*32+c];
  out[(size_t)n*32+o]=acc;
}

// ---------------- host driver ----------------
extern "C" void kernel_launch(void* const* d_in, const int* in_sizes, int n_in,
                              void* d_out, int out_size, void* d_ws, size_t ws_size,
                              hipStream_t stream){
  (void)in_sizes; (void)n_in; (void)out_size; (void)ws_size;
  const float* node_attr=(const float*)d_in[0];
  const float* edge_attr=(const float*)d_in[1];
  const float* pos      =(const float*)d_in[2];
  const int*   src      =(const int*)  d_in[3];
  const int*   dst      =(const int*)  d_in[4];
  // params in setup_inputs() dict insertion order:
  const float* l1kv_w1=(const float*)d_in[5];  const float* l1kv_b1=(const float*)d_in[6];
  const float* l1kv_w2=(const float*)d_in[7];  const float* l1kv_b2=(const float*)d_in[8];
  const float* l1kv_w3=(const float*)d_in[9];  const float* l1kv_b3=(const float*)d_in[10];
  const float* l2kv_w1=(const float*)d_in[11]; const float* l2kv_b1=(const float*)d_in[12];
  const float* l2kv_w2=(const float*)d_in[13]; const float* l2kv_b2=(const float*)d_in[14];
  const float* l2kv_w3=(const float*)d_in[15]; const float* l2kv_b3=(const float*)d_in[16];
  const float* fin_w1 =(const float*)d_in[17]; const float* fin_b1 =(const float*)d_in[18];
  const float* fin_w2 =(const float*)d_in[19]; const float* fin_b2 =(const float*)d_in[20];
  const float* fin_w3 =(const float*)d_in[21]; const float* fin_b3 =(const float*)d_in[22];
  const float* l1_wq0 =(const float*)d_in[23]; const float* l1_ws0 =(const float*)d_in[24];
  const float *l1_wo[3],*l1_ng[3],*l1_nb[3],*l2_wq[3],*l2_wo[3],*l2_ws[3],*l2_ng[3],*l2_nb[3];
  for(int d=0;d<3;++d){
    l1_wo[d]=(const float*)d_in[25+8*d]; l1_ng[d]=(const float*)d_in[26+8*d];
    l1_nb[d]=(const float*)d_in[27+8*d]; l2_wq[d]=(const float*)d_in[28+8*d];
    l2_wo[d]=(const float*)d_in[29+8*d]; l2_ws[d]=(const float*)d_in[30+8*d];
    l2_ng[d]=(const float*)d_in[31+8*d]; l2_nb[d]=(const float*)d_in[32+8*d];
  }
  const float* fin_ws0=(const float*)d_in[49];
  float* out=(float*)d_out;
  float* ws =(float*)d_ws;

  float* cg   = ws+WS_CG;   float* Y    = ws+WS_Y;    float* esc  = ws+WS_ESC;
  float* basis= ws+WS_BASIS;
  float* h2a  = ws+WS_H2A;  float* h2b  = ws+WS_H2B;  float* h2c  = ws+WS_H2C;
  float* kv1  = ws+WS_KV1;  float* kv2  = ws+WS_KV2;
  float* q1   = ws+WS_Q1;   float* q2   = ws+WS_Q2;
  float* logit= ws+WS_LOGIT; unsigned* zmax=(unsigned*)(ws+WS_ZMAX);
  float* aexp = ws+WS_AEXP; float* denom= ws+WS_DENOM;
  float* agg1 = ws+WS_AGG1; float* agg2 = ws+WS_AGG2;
  float* F1   = ws+WS_F1;   float* F2   = ws+WS_F2;
  bf16_t* w3p1=(bf16_t*)(ws+WS_W3P1);
  bf16_t* w3p2=(bf16_t*)(ws+WS_W3P2);
  bf16_t* w3pf=(bf16_t*)(ws+WS_W3PF);
  float* b3p1 = ws+WS_B3P1; float* b3p2 = ws+WS_B3P2; float* b3pf = ws+WS_B3PF;

  auto kvp=[&](float* base,int d){ return base + (size_t)N_EDGES*16*d*d; };
  auto fpp=[&](float* base,int d){ return base + (size_t)N_NODES*32*d*d; };
  auto qpp=[&](float* base,int d){ return base + (size_t)N_NODES*8*d*d; };
  auto app=[&](float* base,int d){ return base + (size_t)N_NODES*8*d*d; };

  // constants (pure, static host buffer -> capturable memcpy node)
  hipMemcpyAsync(cg, g_cg.v, 1225*sizeof(float), hipMemcpyHostToDevice, stream);

  // weight packing + geometry + radial hidden (independent of layer state)
  pack_w3_kernel<<<(768 +255)/256,256,0,stream>>>(l1kv_w3, l1kv_b3, w3p1, b3p1, 720,  15, 16, 768);
  pack_w3_kernel<<<(9728+255)/256,256,0,stream>>>(l2kv_w3, l2kv_b3, w3p2, b3p2, 9728, 32, 32, 9728);
  pack_w3_kernel<<<(3072+255)/256,256,0,stream>>>(fin_w3,  fin_b3,  w3pf, b3pf, 3072, 32, 32, 3072);
  geom_kernel <<<(N_EDGES+255)/256,256,0,stream>>>(pos, edge_attr, src, dst, Y, esc);
  basis_kernel<<<(N_EDGES+127)/128,128,0,stream>>>(Y, cg, basis);
  radial_hidden_kernel<<<N_EDGES/256,256,0,stream>>>(esc,l1kv_w1,l1kv_b1,l1kv_w2,l1kv_b2,h2a);
  radial_hidden_kernel<<<N_EDGES/256,256,0,stream>>>(esc,l2kv_w1,l2kv_b1,l2kv_w2,l2kv_b2,h2b);
  radial_hidden_kernel<<<N_EDGES/256,256,0,stream>>>(esc,fin_w1,fin_b1,fin_w2,fin_b2,h2c);

  // ---- layer 1 ----
  q_kernel<<<(N_NODES*8+255)/256,256,0,stream>>>(node_attr, l1_wq0, q1, L0_IN, 1);
  fused_conv_kernel<<<N_EDGES/16,128,0,stream>>>(h2a, w3p1, b3p1, basis,
      node_attr, nullptr, nullptr, src, dst, kvp(kv1,0), kvp(kv1,1), kvp(kv1,2),
      /*n_din*/1, /*cin*/L0_IN, /*cinP*/16, /*n_dout*/3, /*cout*/16, /*ncolP*/768, /*atomic*/0);
  hipMemsetAsync(zmax, 0, (size_t)N_NODES*4*sizeof(unsigned), stream);
  hipMemsetAsync(denom,0, (size_t)N_NODES*4*sizeof(float), stream);
  logits_kernel<<<(N_EDGES*4+255)/256,256,0,stream>>>(q1, kv1, dst, logit, zmax, 1, 1.f/sqrtf(2.f));
  expsum_kernel<<<(N_EDGES*4+255)/256,256,0,stream>>>(logit, zmax, dst, aexp, denom);
  hipMemsetAsync(agg1, 0, (size_t)N_NODES*8*9*sizeof(float), stream);
  agg_kernel<<<(N_EDGES*8+255)/256,256,0,stream>>>(kv1, aexp, denom, dst, agg1);
  for(int d=0;d<3;++d)
    node_update_kernel<<<N_NODES,32,0,stream>>>(app(agg1,d), (d==0)?node_attr:nullptr,
        l1_wo[d], (d==0)?l1_ws0:nullptr, l1_ng[d], l1_nb[d], fpp(F1,d), d, L0_IN);

  // ---- layer 2 ----
  for(int d=0;d<3;++d)
    q_kernel<<<(N_NODES*8*(2*d+1)+255)/256,256,0,stream>>>(fpp(F1,d), l2_wq[d], qpp(q2,d), 32, 2*d+1);
  fused_conv_kernel<<<N_EDGES/16,128,0,stream>>>(h2b, w3p2, b3p2, basis,
      fpp(F1,0), fpp(F1,1), fpp(F1,2), src, dst, kvp(kv2,0), kvp(kv2,1), kvp(kv2,2),
      3, 32, 32, 3, 16, 9728, 0);
  hipMemsetAsync(zmax, 0, (size_t)N_NODES*4*sizeof(unsigned), stream);
  hipMemsetAsync(denom,0, (size_t)N_NODES*4*sizeof(float), stream);
  logits_kernel<<<(N_EDGES*4+255)/256,256,0,stream>>>(q2, kv2, dst, logit, zmax, 3, 1.f/sqrtf(18.f));
  expsum_kernel<<<(N_EDGES*4+255)/256,256,0,stream>>>(logit, zmax, dst, aexp, denom);
  hipMemsetAsync(agg2, 0, (size_t)N_NODES*8*9*sizeof(float), stream);
  agg_kernel<<<(N_EDGES*8+255)/256,256,0,stream>>>(kv2, aexp, denom, dst, agg2);
  for(int d=0;d<3;++d)
    node_update_kernel<<<N_NODES,32,0,stream>>>(app(agg2,d), fpp(F1,d),
        l2_wo[d], l2_ws[d], l2_ng[d], l2_nb[d], fpp(F2,d), d, 32);

  // ---- final conv + aggregation ----
  out_self_kernel<<<(N_NODES*32+255)/256,256,0,stream>>>(fpp(F2,0), fin_ws0, out);
  fused_conv_kernel<<<N_EDGES/16,128,0,stream>>>(h2c, w3pf, b3pf, basis,
      fpp(F2,0), fpp(F2,1), fpp(F2,2), src, dst, out, nullptr, nullptr,
      3, 32, 32, 1, L0_OUT, 3072, /*atomic*/1);
}
